// GAT_82738249990833
// MI455X (gfx1250) — matrix operand order repaired and checked by reading, not compiled, not run on hardware
//
#include <hip/hip_runtime.h>
#include <hip/hip_bf16.h>
#include <math.h>

typedef __bf16 bf16;
typedef __attribute__((ext_vector_type(16))) __bf16 v16bf;
typedef __attribute__((ext_vector_type(8)))  __bf16 v8bf;
typedef __attribute__((ext_vector_type(8)))  float  v8f;

#define GAT_N    50000
#define GAT_D    16
#define GAT_IN   256
#define GAT_OUT  128
#define GAT_K    4
#define GAT_NOUT (GAT_K * GAT_OUT)   // 512
#define SLOPE    0.01f

// ---------------- workspace layout (bytes) ----------------
// Xbf : N*IN bf16            = 25,600,000
// Wbf : NOUT*IN bf16         =    262,144
// Wh  : N*NOUT f32           = 102,400,000
// eN  : N*K f32              =    800,000
// eS  : N*K f32              =    800,000
#define OFF_XBF 0
#define OFF_WBF 25600000
#define OFF_WH  25862144
#define OFF_EN  128262144
#define OFF_ES  129062144

// -------------------------------------------------------------------
// fp32 -> bf16 pack, 8 elements per thread (counts divisible by 8)
// -------------------------------------------------------------------
__global__ __launch_bounds__(256) void to_bf16_v(const float* __restrict__ src,
                                                 bf16* __restrict__ dst,
                                                 int count8) {
    int i = blockIdx.x * blockDim.x + threadIdx.x;
    if (i >= count8) return;
    const float4 f0 = ((const float4*)src)[i * 2 + 0];
    const float4 f1 = ((const float4*)src)[i * 2 + 1];
    v8bf o;
    o[0] = (bf16)f0.x; o[1] = (bf16)f0.y; o[2] = (bf16)f0.z; o[3] = (bf16)f0.w;
    o[4] = (bf16)f1.x; o[5] = (bf16)f1.y; o[6] = (bf16)f1.z; o[7] = (bf16)f1.w;
    *(v8bf*)(dst + (size_t)i * 8) = o;
}

// -------------------------------------------------------------------
// Wh(N x 512) = Xbf(N x 256) * Wbf(512 x 256)^T via v_wmma_f32_16x16x32_bf16
// One 16x16 C tile per wave; 8 waves (256 thr) per block.
// A-frag layout (16-bit A 16x32): lanes 0-15 row M=lane hold K=0..7 (V0..3)
//   and K=16..23 (V4..7); lanes 16-31 hold K=8..15 / K=24..31  -> two 16B loads.
// B-frag layout (16-bit B 32x16): lane 0-15 col N=lane, K=0..7 / K=16..23;
//   lanes 16-31 = K 8..15 / K 24..31 -> two 16B loads from Wbf row.
// -------------------------------------------------------------------
__global__ __launch_bounds__(256) void wh_gemm(const bf16* __restrict__ Xbf,
                                               const bf16* __restrict__ Wbf,
                                               float* __restrict__ Wh) {
    const int wave = blockIdx.x * 8 + (threadIdx.x >> 5);
    const int lane = threadIdx.x & 31;
    const int tn   = (wave & 31) << 4;   // output-col tile (0..511)
    const int tm   = (wave >> 5) << 4;   // node-row tile
    const int half = lane >> 4;          // 0 | 1
    const int l16  = lane & 15;

    const bf16* arow = Xbf + (size_t)(tm + l16) * GAT_IN + half * 8;
    const bf16* brow = Wbf + (size_t)(tn + l16) * GAT_IN + half * 8;

    v8f c = {};
#pragma unroll
    for (int kk = 0; kk < GAT_IN; kk += 32) {
        const v8bf alo = *(const v8bf*)(arow + kk);        // K = base..base+7
        const v8bf ahi = *(const v8bf*)(arow + kk + 16);   // K = base+16..+23
        v16bf afrag;
#pragma unroll
        for (int i = 0; i < 8; ++i) { afrag[i] = alo[i]; afrag[i + 8] = ahi[i]; }
        const v8bf blo = *(const v8bf*)(brow + kk);
        const v8bf bhi = *(const v8bf*)(brow + kk + 16);
        v16bf bfrag;
#pragma unroll
        for (int i = 0; i < 8; ++i) { bfrag[i] = blo[i]; bfrag[i + 8] = bhi[i]; }
        c = __builtin_amdgcn_wmma_f32_16x16x32_bf16(
                false, afrag, false, bfrag, (short)0, c, false, false);
    }

    // C: VGPR r -> lanes0-15 row tm+r, lanes16-31 row tm+8+r; col tn+l16
    float* crow = Wh + (size_t)(tm + half * 8) * GAT_NOUT + tn + l16;
#pragma unroll
    for (int r = 0; r < 8; ++r) crow[(size_t)r * GAT_NOUT] = c[r];
}

// -------------------------------------------------------------------
// eN[n,k] = Wh[n, k*128 : ] . a[k, 0:128] ; eS uses a[k, 128:256]
// One wave per node, 4 nodes per 128-thread block.
// -------------------------------------------------------------------
__global__ __launch_bounds__(128) void e_compute(const float* __restrict__ Wh,
                                                 const float* __restrict__ a,
                                                 float* __restrict__ eN,
                                                 float* __restrict__ eS) {
    const int n    = blockIdx.x * 4 + (threadIdx.x >> 5);
    const int lane = threadIdx.x & 31;
    const int c0   = lane * 4;
#pragma unroll
    for (int k = 0; k < GAT_K; ++k) {
        const float4 v   = *(const float4*)(Wh + (size_t)n * GAT_NOUT + k * GAT_OUT + c0);
        const float4 anb = *(const float4*)(a + k * (2 * GAT_OUT) + c0);
        const float4 asf = *(const float4*)(a + k * (2 * GAT_OUT) + GAT_OUT + c0);
        float snb = v.x * anb.x + v.y * anb.y + v.z * anb.z + v.w * anb.w;
        float ssf = v.x * asf.x + v.y * asf.y + v.z * asf.z + v.w * asf.w;
#pragma unroll
        for (int m = 16; m >= 1; m >>= 1) {
            snb += __shfl_xor(snb, m, 32);
            ssf += __shfl_xor(ssf, m, 32);
        }
        if (lane == 0) {
            eN[n * GAT_K + k] = snb;
            eS[n * GAT_K + k] = ssf;
        }
    }
}

// -------------------------------------------------------------------
// Per node: leaky-relu + softmax over 16 neighbor scores, then
// out[n, k*128+o] = sum_d alpha_d * Wh[idx[n,d], k*128+o].
// 128 threads per block: wave = head, lane covers 4 channels (float4).
// -------------------------------------------------------------------
__global__ __launch_bounds__(128) void attn_gather(const float* __restrict__ Wh,
                                                   const float* __restrict__ eN,
                                                   const float* __restrict__ eS,
                                                   const int* __restrict__ idx,
                                                   float* __restrict__ out) {
    const int n    = blockIdx.x;
    const int k    = threadIdx.x >> 5;
    const int lane = threadIdx.x & 31;

    const float es = eS[n * GAT_K + k];
    int   nb[GAT_D];
    float e[GAT_D];
#pragma unroll
    for (int d = 0; d < GAT_D; ++d) {
        nb[d] = idx[n * GAT_D + d];
        float v = eN[nb[d] * GAT_K + k] + es;
        e[d] = (v > 0.0f) ? v : SLOPE * v;
    }
    float mx = e[0];
#pragma unroll
    for (int d = 1; d < GAT_D; ++d) mx = fmaxf(mx, e[d]);
    float s = 0.0f;
#pragma unroll
    for (int d = 0; d < GAT_D; ++d) { e[d] = __expf(e[d] - mx); s += e[d]; }
    const float inv = 1.0f / s;

    const float* base = Wh + k * GAT_OUT + lane * 4;
    float4 acc = make_float4(0.f, 0.f, 0.f, 0.f);
#pragma unroll
    for (int d = 0; d < GAT_D; ++d) {
        const float4 v = *(const float4*)(base + (size_t)nb[d] * GAT_NOUT);
        const float al = e[d] * inv;
        acc.x = fmaf(al, v.x, acc.x);
        acc.y = fmaf(al, v.y, acc.y);
        acc.z = fmaf(al, v.z, acc.z);
        acc.w = fmaf(al, v.w, acc.w);
    }
    *(float4*)(out + (size_t)n * GAT_NOUT + k * GAT_OUT + lane * 4) = acc;
}

// -------------------------------------------------------------------
extern "C" void kernel_launch(void* const* d_in, const int* in_sizes, int n_in,
                              void* d_out, int out_size, void* d_ws, size_t ws_size,
                              hipStream_t stream) {
    const float* X   = (const float*)d_in[0];   // (N, 256)
    const float* W   = (const float*)d_in[1];   // (4, 128, 256)  == Wbf source layout
    const float* a   = (const float*)d_in[2];   // (4, 256)
    const int*   idx = (const int*)d_in[3];     // (N, 16)

    char* ws = (char*)d_ws;
    bf16*  Xbf = (bf16*)(ws + OFF_XBF);
    bf16*  Wbf = (bf16*)(ws + OFF_WBF);
    float* Wh  = (float*)(ws + OFF_WH);
    float* eN  = (float*)(ws + OFF_EN);
    float* eS  = (float*)(ws + OFF_ES);

    // X: 12,800,000 elems -> 1,600,000 groups of 8; W: 131,072 -> 16,384
    to_bf16_v<<<(1600000 + 255) / 256, 256, 0, stream>>>(X, Xbf, 1600000);
    to_bf16_v<<<(16384 + 255) / 256, 256, 0, stream>>>(W, Wbf, 16384);

    // 3125 row-tiles * 32 col-tiles = 100000 wave-tiles / 8 waves per block
    wh_gemm<<<12500, 256, 0, stream>>>(Xbf, Wbf, Wh);

    e_compute<<<GAT_N / 4, 128, 0, stream>>>(Wh, a, eN, eS);

    attn_gather<<<GAT_N, 128, 0, stream>>>(Wh, eN, eS, idx, (float*)d_out);
}
